// HydraBlock_80032420593968
// MI455X (gfx1250) — matrix operand order repaired and checked by reading, not compiled, not run on hardware
//
#include <hip/hip_runtime.h>
#include <hip/hip_bf16.h>
#include <math.h>

#define D_MODEL 128
#define N_HEADS 8
#define HEAD_DIM 16
#define N_EXPERTS 256
#define BB 4
#define TT 512
#define MTOK (BB * TT)   // 2048 tokens

typedef __attribute__((ext_vector_type(16))) _Float16 v16h;
typedef __attribute__((ext_vector_type(8)))  float    v8f;

// ---------------------------------------------------------------------------
// WMMA helpers (wave32, 16x16x32 f16 -> f32)
// ---------------------------------------------------------------------------
__device__ __forceinline__ v8f wmma_f16(v16h a, v16h b, v8f c) {
    return __builtin_amdgcn_wmma_f32_16x16x32_f16(
        /*neg_a=*/false, a, /*neg_b=*/false, b,
        /*c_mod=*/(short)0, c, /*reuse_a=*/false, /*reuse_b=*/false);
}

// A-fragment: 16x32 f16 tile, row-major [16][ld].
// ISA layout (16-bit A 16x32): lane l (0..15): M=l, K={0..7,16..23};
// lane l+16: M=l, K={8..15,24..31}. Two contiguous 16B loads per lane.
__device__ __forceinline__ v16h load_fragA(const _Float16* p0, int ld) {
    int l  = threadIdx.x & 31;
    const _Float16* p = p0 + (l & 15) * ld + ((l >> 4) << 3);
    v16h f;
    *reinterpret_cast<float4*>(&f)       = *reinterpret_cast<const float4*>(p);
    *(reinterpret_cast<float4*>(&f) + 1) = *reinterpret_cast<const float4*>(p + 16);
    return f;
}

// B-fragment: B is 32(K)x16(N); we read from row-major Bt[N][ld] (i.e. B^T).
// ISA layout (same K order as A): lane l (0..15): n=l, K={0..7,16..23};
// lane l+16: n=l, K={8..15,24..31}. Two contiguous 16B loads per lane.
__device__ __forceinline__ v16h load_fragB(const _Float16* p0, int ld) {
    int l  = threadIdx.x & 31;
    const _Float16* p = p0 + (l & 15) * ld + ((l >> 4) << 3);
    v16h f;
    *reinterpret_cast<float4*>(&f)       = *reinterpret_cast<const float4*>(p);
    *(reinterpret_cast<float4*>(&f) + 1) = *reinterpret_cast<const float4*>(p + 16);
    return f;
}

// ---------------------------------------------------------------------------
// Generic WMMA GEMM: C[M,N] = A[M,K](f16) * Bt[N,K](f16)^T + bias[N]
// 1 wave per 16x16 tile, 8 waves per block. Optional f32 and f16 outputs.
// ---------------------------------------------------------------------------
__global__ void k_gemm_wmma(const _Float16* __restrict__ A,
                            const _Float16* __restrict__ Bt,
                            const float* __restrict__ bias,
                            float* __restrict__ C, _Float16* __restrict__ Ch,
                            int M, int N, int K) {
    int tilesN = N >> 4;
    int tile = blockIdx.x * 8 + (threadIdx.x >> 5);
    if (tile >= (M >> 4) * tilesN) return;
    int tm = (tile / tilesN) << 4;
    int tn = (tile % tilesN) << 4;

    // Row prefetch (lowers to global_prefetch_b8): lanes 0-15 -> A rows,
    // lanes 16-31 -> B rows.
    {
        int l = threadIdx.x & 31;
        if (l < 16) __builtin_prefetch(A  + (size_t)(tm + l) * K, 0, 0);
        else        __builtin_prefetch(Bt + (size_t)(tn + (l - 16)) * K, 0, 0);
    }

    v8f acc = {};
    for (int k0 = 0; k0 < K; k0 += 32) {
        v16h a = load_fragA(A  + (size_t)tm * K + k0, K);
        v16h b = load_fragB(Bt + (size_t)tn * K + k0, K);
        acc = wmma_f16(a, b, acc);
    }
    int l  = threadIdx.x & 31;
    int nl = l & 15;
    int mb = (l >> 4) << 3;           // lanes 0-15 -> M=r, lanes 16-31 -> M=8+r
    float bv = bias ? bias[tn + nl] : 0.0f;
#pragma unroll
    for (int r = 0; r < 8; ++r) {
        int m = tm + mb + r;
        float v = acc[r] + bv;
        if (C)  C [(size_t)m * N + tn + nl] = v;
        if (Ch) Ch[(size_t)m * N + tn + nl] = (_Float16)v;
    }
}

// ---------------------------------------------------------------------------
// Transpose+convert weight: W[K,N] f32 -> Wt[N,K] f16
// ---------------------------------------------------------------------------
__global__ void k_transpose_h(const float* __restrict__ W, _Float16* __restrict__ Wt,
                              int K, int N) {
    int i = blockIdx.x * blockDim.x + threadIdx.x;
    if (i >= K * N) return;
    int k = i / N, n = i % N;
    Wt[(size_t)n * K + k] = (_Float16)W[i];
}

// ---------------------------------------------------------------------------
// RMSNorm per row of 128, write f16 and/or f32
// ---------------------------------------------------------------------------
__global__ void k_rmsnorm(const float* __restrict__ x, const float* __restrict__ gamma,
                          _Float16* __restrict__ oh, float* __restrict__ of) {
    int row = blockIdx.x, t = threadIdx.x;
    float v = x[(size_t)row * D_MODEL + t];
    __shared__ float red[D_MODEL];
    red[t] = v * v; __syncthreads();
    for (int s = 64; s > 0; s >>= 1) { if (t < s) red[t] += red[t + s]; __syncthreads(); }
    float norm = sqrtf(red[0] / (float)D_MODEL + 1e-6f);
    float o = gamma[t] * (v / norm);
    if (oh) oh[(size_t)row * D_MODEL + t] = (_Float16)o;
    if (of) of[(size_t)row * D_MODEL + t] = o;
}

// ---------------------------------------------------------------------------
// RoPE (NeoX half-split, head_dim 16) + elu()+1, in-place on q and k
// ---------------------------------------------------------------------------
__device__ __forceinline__ float elu1(float z) { return z > 0.0f ? z + 1.0f : expf(z); }

__global__ void k_rope_elu(float* __restrict__ q, float* __restrict__ k) {
    int idx = blockIdx.x * blockDim.x + threadIdx.x;   // B*T*H*8
    if (idx >= BB * TT * N_HEADS * 8) return;
    int j = idx & 7;
    int rest = idx >> 3;
    int h = rest & (N_HEADS - 1);
    int rest2 = rest >> 3;
    int t = rest2 % TT;
    size_t base = ((size_t)rest2 * N_HEADS + h) * HEAD_DIM;
    float inv = powf(10000.0f, -(float)j / 8.0f);
    float ang = (float)t * inv;
    float s = sinf(ang), c = cosf(ang);
    float q1 = q[base + j], q2 = q[base + j + 8];
    q[base + j]     = elu1(q1 * c - q2 * s);
    q[base + j + 8] = elu1(q1 * s + q2 * c);
    float k1 = k[base + j], k2 = k[base + j + 8];
    k[base + j]     = elu1(k1 * c - k2 * s);
    k[base + j + 8] = elu1(k1 * s + k2 * c);
}

// ---------------------------------------------------------------------------
// Lightning attention context: per (b,h), ctx[16][16] = K^T V over T=512,
// k_sum[16] via WMMA against an all-ones B fragment. One wave per block.
// ---------------------------------------------------------------------------
__global__ void k_ctx(const float* __restrict__ k, const float* __restrict__ v,
                      float* __restrict__ ctx, float* __restrict__ ksum) {
    __shared__ _Float16 kT[16][32];   // [d][t] transposed tile (A operand)
    __shared__ _Float16 vT[16][32];   // [e][t] transposed tile (B^T operand)
    int bh = blockIdx.x;              // 0..31
    int b = bh / N_HEADS, h = bh % N_HEADS;
    int l = threadIdx.x;              // 0..31

    v8f accC = {}, accS = {};
    v16h ones;
#pragma unroll
    for (int i = 0; i < 16; ++i) ones[i] = (_Float16)1.0f;

    for (int t0 = 0; t0 < TT; t0 += 32) {
        int t = t0 + l;
        const float* kp = k + (((size_t)b * TT + t) * N_HEADS + h) * HEAD_DIM;
        const float* vp = v + (((size_t)b * TT + t) * N_HEADS + h) * HEAD_DIM;
        __syncthreads();
#pragma unroll
        for (int d = 0; d < 16; ++d) { kT[d][l] = (_Float16)kp[d]; vT[d][l] = (_Float16)vp[d]; }
        __syncthreads();
        v16h a  = load_fragA(&kT[0][0], 32);   // A: 16(d) x 32(t)
        v16h bb = load_fragB(&vT[0][0], 32);   // B: 32(t) x 16(e) via V^T rows
        accC = wmma_f16(a, bb, accC);          // ctx[d][e] += sum_t k[t,d]*v[t,e]
        accS = wmma_f16(a, ones, accS);        // ksum[d] replicated in every column
    }
    int nl = l & 15, mb = (l >> 4) << 3;
    float* cd = ctx + (size_t)bh * 256;
#pragma unroll
    for (int r = 0; r < 8; ++r) cd[(mb + r) * 16 + nl] = accC[r];
    if (nl == 0) {
        float* sd = ksum + (size_t)bh * 16;
#pragma unroll
        for (int r = 0; r < 8; ++r) sd[mb + r] = accS[r];
    }
}

// ---------------------------------------------------------------------------
// out[t,e] = (q[t,:] @ ctx)[e] / (q[t,:]·k_sum + 1e-6); writes f16 (B,T,128)
// ---------------------------------------------------------------------------
__global__ void k_attn_out(const float* __restrict__ q, const float* __restrict__ ctx,
                           const float* __restrict__ ksum, _Float16* __restrict__ out) {
    int bh = blockIdx.x;
    int b = bh / N_HEADS, h = bh % N_HEADS;
    __shared__ float sc[256];
    __shared__ float sk[16];
    sc[threadIdx.x] = ctx[(size_t)bh * 256 + threadIdx.x];
    if (threadIdx.x < 16) sk[threadIdx.x] = ksum[(size_t)bh * 16 + threadIdx.x];
    __syncthreads();
    int e  = threadIdx.x & 15;
    int tl = threadIdx.x >> 4;
    for (int tc = 0; tc < TT; tc += 16) {
        int t = tc + tl;
        const float* qp = q + (((size_t)b * TT + t) * N_HEADS + h) * HEAD_DIM;
        float s = 0.0f, den = 0.0f;
#pragma unroll
        for (int d = 0; d < 16; ++d) { float qv = qp[d]; s += qv * sc[d * 16 + e]; den += qv * sk[d]; }
        out[((size_t)b * TT + t) * D_MODEL + h * HEAD_DIM + e] = (_Float16)(s / (den + 1e-6f));
    }
}

// ---------------------------------------------------------------------------
// turbo_quant rowwise epilogue on y (rot output): STE forward == quantized
// ---------------------------------------------------------------------------
__global__ void k_turbo_quant(const float* __restrict__ y, const float* __restrict__ scale,
                              _Float16* __restrict__ oh) {
    int row = blockIdx.x, t = threadIdx.x;
    float v = y[(size_t)row * D_MODEL + t];
    __shared__ float red[D_MODEL];
    red[t] = v * v; __syncthreads();
    for (int s = 64; s > 0; s >>= 1) { if (t < s) red[t] += red[t + s]; __syncthreads(); }
    float mag = sqrtf(red[0] / (float)D_MODEL + 1e-6f);
    float quant = rintf((v / mag) * 8.0f) / 8.0f;   // round-half-even like jnp.round
    oh[(size_t)row * D_MODEL + t] = (_Float16)(quant * mag * scale[t]);
}

// ---------------------------------------------------------------------------
// x2 = x + gate*sigmoid(gate)*lin   (elementwise)
// ---------------------------------------------------------------------------
__global__ void k_swiglu_add(const float* __restrict__ x, const float* __restrict__ g,
                             const float* __restrict__ lin, float* __restrict__ x2, int n) {
    int i = blockIdx.x * blockDim.x + threadIdx.x;
    if (i >= n) return;
    float gv = g[i];
    x2[i] = x[i] + gv * (1.0f / (1.0f + expf(-gv))) * lin[i];
}

// ---------------------------------------------------------------------------
// Softmax over 256 experts + top-2 (+renormalized weights), per token
// ---------------------------------------------------------------------------
__global__ void k_topk(const float* __restrict__ scores, int* __restrict__ idx,
                       float* __restrict__ w) {
    int tkn = blockIdx.x, t = threadIdx.x;
    __shared__ float red[N_EXPERTS];
    __shared__ int   ridx[N_EXPERTS];
    float s = scores[(size_t)tkn * N_EXPERTS + t];
    red[t] = s; __syncthreads();
    for (int st = 128; st > 0; st >>= 1) { if (t < st) red[t] = fmaxf(red[t], red[t + st]); __syncthreads(); }
    float mx = red[0]; __syncthreads();
    float e = expf(s - mx);
    red[t] = e; __syncthreads();
    for (int st = 128; st > 0; st >>= 1) { if (t < st) red[t] += red[t + st]; __syncthreads(); }
    float p = e / red[0]; __syncthreads();
    // top-1 (ties -> lowest index, matching lax.top_k)
    red[t] = p; ridx[t] = t; __syncthreads();
    for (int st = 128; st > 0; st >>= 1) {
        if (t < st && red[t + st] > red[t]) { red[t] = red[t + st]; ridx[t] = ridx[t + st]; }
        __syncthreads();
    }
    float v0 = red[0]; int i0 = ridx[0]; __syncthreads();
    red[t] = (t == i0) ? -1.0f : p; ridx[t] = t; __syncthreads();
    for (int st = 128; st > 0; st >>= 1) {
        if (t < st && red[t + st] > red[t]) { red[t] = red[t + st]; ridx[t] = ridx[t + st]; }
        __syncthreads();
    }
    if (t == 0) {
        float v1 = red[0]; int i1 = ridx[0];
        float ws = v0 + v1 + 1e-6f;
        idx[tkn * 2] = i0; idx[tkn * 2 + 1] = i1;
        w[tkn * 2] = v0 / ws; w[tkn * 2 + 1] = v1 / ws;
    }
}

// ---------------------------------------------------------------------------
// Gathered top-2 expert matvecs: y_e = h @ exp_w[e], wa = w0*y0 + w1*y1 (f16)
// ---------------------------------------------------------------------------
__global__ void k_expert(const float* __restrict__ h, const float* __restrict__ exp_w,
                         const int* __restrict__ idx, const float* __restrict__ w,
                         float* __restrict__ y01, _Float16* __restrict__ wah) {
    int tkn = blockIdx.x, o = threadIdx.x;
    __shared__ float hs[D_MODEL];
    hs[o] = h[(size_t)tkn * D_MODEL + o];
    __syncthreads();
    int i0 = idx[tkn * 2], i1 = idx[tkn * 2 + 1];
    const float* W0 = exp_w + (size_t)i0 * D_MODEL * D_MODEL;
    const float* W1 = exp_w + (size_t)i1 * D_MODEL * D_MODEL;
    __builtin_prefetch(W0 + o, 0, 0);
    __builtin_prefetch(W1 + o, 0, 0);
    float a0 = 0.0f, a1 = 0.0f;
    for (int d = 0; d < D_MODEL; ++d) {
        float hv = hs[d];
        a0 += hv * W0[(size_t)d * D_MODEL + o];
        a1 += hv * W1[(size_t)d * D_MODEL + o];
    }
    y01[(size_t)tkn * 2 * D_MODEL + o]           = a0;
    y01[(size_t)tkn * 2 * D_MODEL + D_MODEL + o] = a1;
    float wa = w[tkn * 2] * a0 + w[tkn * 2 + 1] * a1;
    wah[(size_t)tkn * D_MODEL + o] = (_Float16)wa;
}

// ---------------------------------------------------------------------------
// Final: moe swiglu, residual add, top-2 weighted variance -> consensus
// ---------------------------------------------------------------------------
__global__ void k_final(const float* __restrict__ x2, const float* __restrict__ g,
                        const float* __restrict__ lin, const float* __restrict__ y01,
                        const float* __restrict__ w, float* __restrict__ out_x,
                        float* __restrict__ out_c) {
    int tkn = blockIdx.x, o = threadIdx.x;
    float gv = g[(size_t)tkn * D_MODEL + o];
    float m  = gv * (1.0f / (1.0f + expf(-gv))) * lin[(size_t)tkn * D_MODEL + o];
    out_x[(size_t)tkn * D_MODEL + o] = x2[(size_t)tkn * D_MODEL + o] + m;
    float y0 = y01[(size_t)tkn * 2 * D_MODEL + o];
    float y1 = y01[(size_t)tkn * 2 * D_MODEL + D_MODEL + o];
    float var = w[tkn * 2] * (y0 - m) * (y0 - m) + w[tkn * 2 + 1] * (y1 - m) * (y1 - m);
    __shared__ float red[D_MODEL];
    red[o] = var; __syncthreads();
    for (int s = 64; s > 0; s >>= 1) { if (o < s) red[o] += red[o + s]; __syncthreads(); }
    if (o == 0) out_c[tkn] = expf(-red[0] / (float)D_MODEL);
}

// ---------------------------------------------------------------------------
extern "C" void kernel_launch(void* const* d_in, const int* in_sizes, int n_in,
                              void* d_out, int out_size, void* d_ws, size_t ws_size,
                              hipStream_t stream) {
    const float* x       = (const float*)d_in[0];
    const float* g1      = (const float*)d_in[1];
    const float* wq      = (const float*)d_in[2];  const float* bq = (const float*)d_in[3];
    const float* wk      = (const float*)d_in[4];  const float* bk = (const float*)d_in[5];
    const float* wv      = (const float*)d_in[6];  const float* bv = (const float*)d_in[7];
    const float* wo      = (const float*)d_in[8];  const float* bo = (const float*)d_in[9];
    const float* rot     = (const float*)d_in[10];
    const float* tqs     = (const float*)d_in[11];
    const float* sw_w1   = (const float*)d_in[12]; const float* sw_b1 = (const float*)d_in[13];
    const float* sw_w2   = (const float*)d_in[14]; const float* sw_b2 = (const float*)d_in[15];
    const float* g2      = (const float*)d_in[16];
    const float* gate_w  = (const float*)d_in[17]; const float* gate_b = (const float*)d_in[18];
    const float* exp_w   = (const float*)d_in[19];
    const float* moe_w1  = (const float*)d_in[20]; const float* moe_b1 = (const float*)d_in[21];
    const float* moe_w2  = (const float*)d_in[22]; const float* moe_b2 = (const float*)d_in[23];

    char* ws = (char*)d_ws;
    size_t off = 0;
    auto alloc = [&](size_t bytes) -> void* {
        void* p = ws + off;
        off = (off + bytes + 255) & ~(size_t)255;
        return p;
    };
    const int D = D_MODEL, M = MTOK, E = N_EXPERTS;

    _Float16* wqT   = (_Float16*)alloc((size_t)D * D * 2);
    _Float16* wkT   = (_Float16*)alloc((size_t)D * D * 2);
    _Float16* wvT   = (_Float16*)alloc((size_t)D * D * 2);
    _Float16* woT   = (_Float16*)alloc((size_t)D * D * 2);
    _Float16* rotT  = (_Float16*)alloc((size_t)D * D * 2);
    _Float16* sw1T  = (_Float16*)alloc((size_t)D * D * 2);
    _Float16* sw2T  = (_Float16*)alloc((size_t)D * D * 2);
    _Float16* gateT = (_Float16*)alloc((size_t)D * E * 2);
    _Float16* m1T   = (_Float16*)alloc((size_t)D * D * 2);
    _Float16* m2T   = (_Float16*)alloc((size_t)D * D * 2);
    _Float16* xn_h  = (_Float16*)alloc((size_t)M * D * 2);
    float*    qf    = (float*)   alloc((size_t)M * D * 4);
    float*    kf    = (float*)   alloc((size_t)M * D * 4);
    float*    vf    = (float*)   alloc((size_t)M * D * 4);
    float*    ctx   = (float*)   alloc((size_t)32 * 256 * 4);
    float*    ksum  = (float*)   alloc((size_t)32 * 16 * 4);
    _Float16* at_h  = (_Float16*)alloc((size_t)M * D * 2);  // attn pre-proj (f16)
    _Float16* ap_h  = (_Float16*)alloc((size_t)M * D * 2);  // attn post-wo (f16)
    float*    y_f   = (float*)   alloc((size_t)M * D * 4);  // rot output
    _Float16* tq_h  = (_Float16*)alloc((size_t)M * D * 2);
    float*    gt_f  = (float*)   alloc((size_t)M * D * 4);  // swiglu gate
    float*    ln_f  = (float*)   alloc((size_t)M * D * 4);  // swiglu linear
    float*    x2_f  = (float*)   alloc((size_t)M * D * 4);
    float*    h_f   = (float*)   alloc((size_t)M * D * 4);
    _Float16* h_h   = (_Float16*)alloc((size_t)M * D * 2);
    float*    sc_f  = (float*)   alloc((size_t)M * E * 4);  // gate scores
    int*      tidx  = (int*)     alloc((size_t)M * 2 * 4);
    float*    tw    = (float*)   alloc((size_t)M * 2 * 4);
    float*    y01   = (float*)   alloc((size_t)M * 2 * D * 4);
    _Float16* wa_h  = (_Float16*)alloc((size_t)M * D * 2);
    float*    g2_f  = (float*)   alloc((size_t)M * D * 4);
    float*    l2_f  = (float*)   alloc((size_t)M * D * 4);

    float* out_x = (float*)d_out;
    float* out_c = (float*)d_out + (size_t)M * D;

    // --- weight transposes (f32 [K,N] -> f16 [N,K]) ---
    int tthr = D * D, tblk = (tthr + 255) / 256;
    k_transpose_h<<<tblk, 256, 0, stream>>>(wq,    wqT,  D, D);
    k_transpose_h<<<tblk, 256, 0, stream>>>(wk,    wkT,  D, D);
    k_transpose_h<<<tblk, 256, 0, stream>>>(wv,    wvT,  D, D);
    k_transpose_h<<<tblk, 256, 0, stream>>>(wo,    woT,  D, D);
    k_transpose_h<<<tblk, 256, 0, stream>>>(rot,   rotT, D, D);
    k_transpose_h<<<tblk, 256, 0, stream>>>(sw_w1, sw1T, D, D);
    k_transpose_h<<<tblk, 256, 0, stream>>>(sw_w2, sw2T, D, D);
    k_transpose_h<<<(D * E + 255) / 256, 256, 0, stream>>>(gate_w, gateT, D, E);
    k_transpose_h<<<tblk, 256, 0, stream>>>(moe_w1, m1T, D, D);
    k_transpose_h<<<tblk, 256, 0, stream>>>(moe_w2, m2T, D, D);

    // --- attention path ---
    k_rmsnorm<<<M, D, 0, stream>>>(x, g1, xn_h, nullptr);
    int gB = (M / 16) * (D / 16) / 8;        // 128 blocks, 8 waves each
    k_gemm_wmma<<<gB, 256, 0, stream>>>(xn_h, wqT, bq, qf, nullptr, M, D, D);
    k_gemm_wmma<<<gB, 256, 0, stream>>>(xn_h, wkT, bk, kf, nullptr, M, D, D);
    k_gemm_wmma<<<gB, 256, 0, stream>>>(xn_h, wvT, bv, vf, nullptr, M, D, D);
    k_rope_elu<<<(BB * TT * N_HEADS * 8 + 255) / 256, 256, 0, stream>>>(qf, kf);
    k_ctx<<<BB * N_HEADS, 32, 0, stream>>>(kf, vf, ctx, ksum);
    k_attn_out<<<BB * N_HEADS, 256, 0, stream>>>(qf, ctx, ksum, at_h);
    k_gemm_wmma<<<gB, 256, 0, stream>>>(at_h, woT, bo, nullptr, ap_h, M, D, D);

    // --- turbo_quant ---
    k_gemm_wmma<<<gB, 256, 0, stream>>>(ap_h, rotT, nullptr, y_f, nullptr, M, D, D);
    k_turbo_quant<<<M, D, 0, stream>>>(y_f, tqs, tq_h);

    // --- swiglu + residual ---
    k_gemm_wmma<<<gB, 256, 0, stream>>>(tq_h, sw1T, sw_b1, gt_f, nullptr, M, D, D);
    k_gemm_wmma<<<gB, 256, 0, stream>>>(tq_h, sw2T, sw_b2, ln_f, nullptr, M, D, D);
    k_swiglu_add<<<(M * D + 255) / 256, 256, 0, stream>>>(x, gt_f, ln_f, x2_f, M * D);

    // --- MoE ---
    k_rmsnorm<<<M, D, 0, stream>>>(x2_f, g2, h_h, h_f);
    int gBE = (M / 16) * (E / 16) / 8;       // 256 blocks
    k_gemm_wmma<<<gBE, 256, 0, stream>>>(h_h, gateT, gate_b, sc_f, nullptr, M, E, D);
    k_topk<<<M, E, 0, stream>>>(sc_f, tidx, tw);
    k_expert<<<M, D, 0, stream>>>(h_f, exp_w, tidx, tw, y01, wa_h);
    k_gemm_wmma<<<gB, 256, 0, stream>>>(wa_h, m1T, moe_b1, g2_f, nullptr, M, D, D);
    k_gemm_wmma<<<gB, 256, 0, stream>>>(wa_h, m2T, moe_b2, l2_f, nullptr, M, D, D);
    k_final<<<M, D, 0, stream>>>(x2_f, g2_f, l2_f, y01, tw, out_x, out_c);

    (void)in_sizes; (void)n_in; (void)out_size; (void)ws_size;
}
